// Bert_BiLSTM_CRF_54554674593907
// MI455X (gfx1250) — compile-verified, matrix operand order checked by reading, not run on hardware
//
#include <hip/hip_runtime.h>
#include <math.h>

#define DEV __device__ __forceinline__

typedef __attribute__((ext_vector_type(16))) __bf16 bf16x16;
typedef __attribute__((ext_vector_type(8)))  __bf16 bf16x8;
typedef __attribute__((ext_vector_type(8)))  float  f32x8;
typedef __attribute__((ext_vector_type(4)))  unsigned int u32x4;
typedef __attribute__((ext_vector_type(8)))  int    i32x8;
typedef __attribute__((ext_vector_type(4)))  int    i32x4;

namespace {
constexpr int   Bz   = 64;
constexpr int   T    = 512;
constexpr int   D    = 768;    // LSTM input dim (both layers: 768 = 2*H)
constexpr int   H    = 384;
constexpr int   G4   = 4 * H;  // 1536 gate width
constexpr int   K    = 9;
constexpr int   R    = T * Bz; // 32768 rows
constexpr float NEGV = -10000.0f;
constexpr int   START_TAG = 7;
}

// ---------------------------------------------------------------------------
// WMMA fragment helpers (bf16 16x16x32, ISA 7.12.2 layouts)
// A/W fragment: lane L holds row (L&15); elements 0..7 = K[kb..kb+7],
// 8..15 = K[kb+16..kb+23], kb = (L&16)?8:0.
// ---------------------------------------------------------------------------
DEV bf16x16 load_frag(const __bf16* __restrict__ base, int ld, int row0, int k0, int lane) {
  const __bf16* p = base + (size_t)(row0 + (lane & 15)) * ld + (k0 + ((lane & 16) ? 8 : 0));
  bf16x8 lo = *(const bf16x8*)(p);
  bf16x8 hi = *(const bf16x8*)(p + 16);
  return __builtin_shufflevector(lo, hi, 0,1,2,3,4,5,6,7,8,9,10,11,12,13,14,15);
}

DEV f32x8 wmma_bf16(bf16x16 a, bf16x16 b, f32x8 c) {
  return __builtin_amdgcn_wmma_f32_16x16x32_bf16(false, a, false, b, (short)0, c, false, false);
}

// C/D tile: VGPR r: lanes 0-15 -> M=m0+r, lanes 16-31 -> M=m0+8+r; N = n0+(lane&15)
DEV void store_tile_bf16(__bf16* __restrict__ Y, int ld, int m0, int n0, int lane, f32x8 c) {
  int col = n0 + (lane & 15);
  int r0  = m0 + ((lane & 16) ? 8 : 0);
#pragma unroll
  for (int r = 0; r < 8; ++r) Y[(size_t)(r0 + r) * ld + col] = (__bf16)c[r];
}
DEV void store_tile_f32(float* __restrict__ Y, int ld, int m0, int n0, int lane, f32x8 c) {
  int col = n0 + (lane & 15);
  int r0  = m0 + ((lane & 16) ? 8 : 0);
#pragma unroll
  for (int r = 0; r < 8; ++r) Y[(size_t)(r0 + r) * ld + col] = c[r];
}

// ---------------------------------------------------------------------------
// Tensor Data Mover: async 2D tile load global -> LDS (bf16 elements).
// Builds D# group0/group1 per CDNA5 ISA:
//   group0: count=1 | lds_addr[63:32] | global_addr[120:64] | type=2[127:126]
//   group1: data_size=1(2B)<<16 | tensor_dim0[79:48] | tensor_dim1[111:80]
//           | tile_dim0[127:112] | tile_dim1[143:128] | dim0_stride[207:160]
// All operands are workgroup-uniform -> SGPR groups as TDM requires.
// ---------------------------------------------------------------------------
DEV void tdm_load_2d_bf16(const void* gaddr, unsigned lds_off,
                          unsigned tile_d0, unsigned tile_d1,
                          unsigned tensor_d0, unsigned tensor_d1,
                          unsigned stride0_elems) {
  unsigned long long ga = (unsigned long long)gaddr;
  u32x4 g0;
  g0[0] = 1u;                                             // count=1, user descriptor
  g0[1] = lds_off;                                        // lds_addr (bytes)
  g0[2] = (unsigned)(ga & 0xffffffffu);                   // global_addr[31:0]
  g0[3] = (unsigned)((ga >> 32) & 0x01ffffffu) | (2u << 30);  // addr[56:32], type=2
  i32x8 g1;
  g1[0] = (int)(1u << 16);                                // data_size=1 (2 bytes)
  g1[1] = (int)((tensor_d0 & 0xffffu) << 16);             // tensor_dim0[15:0]
  g1[2] = (int)((tensor_d0 >> 16) | ((tensor_d1 & 0xffffu) << 16));
  g1[3] = (int)((tensor_d1 >> 16) | (tile_d0 << 16));     // tile_dim0
  g1[4] = (int)(tile_d1 & 0xffffu);                       // tile_dim1 (tile_dim2=0)
  g1[5] = (int)stride0_elems;                             // tensor_dim0_stride[31:0]
  g1[6] = 0;
  g1[7] = 0;
  i32x4 gz = {0, 0, 0, 0};
#if defined(__clang_major__) && (__clang_major__ >= 23)
  i32x8 gz8 = {0, 0, 0, 0, 0, 0, 0, 0};
  __builtin_amdgcn_tensor_load_to_lds(g0, g1, gz, gz, gz8, 0);
#else
  __builtin_amdgcn_tensor_load_to_lds(g0, g1, gz, gz, 0);
#endif
}

DEV unsigned lds_offset_of(const void* p) {
  return (unsigned)(unsigned long long)(uintptr_t)p;      // LDS aperture: addr[31:0] = LDS offset
}

// ---------------------------------------------------------------------------
// Conversion kernels
// ---------------------------------------------------------------------------
__global__ void k_cvt_bf16(const float* __restrict__ src, __bf16* __restrict__ dst, int n) {
  int i = blockIdx.x * blockDim.x + threadIdx.x;
  if (i < n) dst[i] = (__bf16)src[i];
}

// embeds [B,T,D] f32 -> x [T*B, D] bf16  (time-major like the reference)
__global__ void k_embed_tr(const float* __restrict__ e, __bf16* __restrict__ x) {
  int i = blockIdx.x * blockDim.x + threadIdx.x;
  if (i >= Bz * T * D) return;
  int d = i % D;
  int bt = i / D;
  int t = bt % T;
  int b = bt / T;
  x[((size_t)(t * Bz + b)) * D + d] = (__bf16)e[i];
}

// ---------------------------------------------------------------------------
// Input projection: Y[dir] = X[R,768] @ Wih[dir]^T  -> [R,1536] bf16
// Block 256 thr = 8 waves; block tile 64(M) x 128(N); K staged in 32-wide
// double-buffered LDS tiles filled by the Tensor Data Mover (wave 0 issues,
// s_wait_tensorcnt retires, barrier publishes). grid = (12, 512, 2).
// ---------------------------------------------------------------------------
__global__ void __launch_bounds__(256) k_xproj(const __bf16* __restrict__ X,
                                               const __bf16* __restrict__ Wf,
                                               const __bf16* __restrict__ Wb,
                                               __bf16* __restrict__ Y) {
  __shared__ alignas(16) __bf16 Abuf[2][64 * 32];    //  8 KB
  __shared__ alignas(16) __bf16 Bbuf[2][128 * 32];   // 16 KB
  const int dir = blockIdx.z;
  const __bf16* W = dir ? Wb : Wf;
  __bf16* Yd = Y + (size_t)dir * R * G4;
  const int lane = threadIdx.x & 31;
  const int wave = threadIdx.x >> 5;
  const int Mblk = blockIdx.y * 64;
  const int Nblk = blockIdx.x * 128;
  const int am0 = (wave & 1) * 32;    // wave's M offset within block tile
  const int bn0 = (wave >> 1) * 32;   // wave's N offset within block tile
  const __bf16* Xblk = X + (size_t)Mblk * D;
  const __bf16* Wblk = W + (size_t)Nblk * D;

  f32x8 c00 = {}, c01 = {}, c10 = {}, c11 = {};

  if (wave == 0) {  // prologue: stage k-block 0
    tdm_load_2d_bf16(Xblk, lds_offset_of(&Abuf[0][0]), 32, 64, D, (unsigned)R, D);
    tdm_load_2d_bf16(Wblk, lds_offset_of(&Bbuf[0][0]), 32, 128, D, (unsigned)G4, D);
  }

  int buf = 0;
  constexpr int KB = D / 32;  // 24 k-blocks
  for (int kb = 0; kb < KB; ++kb) {
    if (wave == 0) {
      if (kb + 1 < KB) {  // stage next k-block into the other buffer
        tdm_load_2d_bf16(Xblk + (kb + 1) * 32, lds_offset_of(&Abuf[buf ^ 1][0]),
                         32, 64, D, (unsigned)R, D);
        tdm_load_2d_bf16(Wblk + (kb + 1) * 32, lds_offset_of(&Bbuf[buf ^ 1][0]),
                         32, 128, D, (unsigned)G4, D);
        __builtin_amdgcn_s_wait_tensorcnt(2);  // current stage done, next in flight
      } else {
        __builtin_amdgcn_s_wait_tensorcnt(0);
      }
    }
    __syncthreads();  // publish staged LDS tiles to all waves

    const __bf16* Ab = Abuf[buf];
    const __bf16* Bb = Bbuf[buf];
    bf16x16 a0 = load_frag(Ab, 32, am0,      0, lane);
    bf16x16 a1 = load_frag(Ab, 32, am0 + 16, 0, lane);
    bf16x16 b0 = load_frag(Bb, 32, bn0,      0, lane);
    bf16x16 b1 = load_frag(Bb, 32, bn0 + 16, 0, lane);
    c00 = wmma_bf16(a0, b0, c00);
    c01 = wmma_bf16(a0, b1, c01);
    c10 = wmma_bf16(a1, b0, c10);
    c11 = wmma_bf16(a1, b1, c11);

    __syncthreads();  // all waves done with `buf^1` consumers before TDM reuses it
    buf ^= 1;
  }

  store_tile_bf16(Yd, G4, Mblk + am0,      Nblk + bn0,      lane, c00);
  store_tile_bf16(Yd, G4, Mblk + am0,      Nblk + bn0 + 16, lane, c01);
  store_tile_bf16(Yd, G4, Mblk + am0 + 16, Nblk + bn0,      lane, c10);
  store_tile_bf16(Yd, G4, Mblk + am0 + 16, Nblk + bn0 + 16, lane, c11);
}

// ---------------------------------------------------------------------------
// Persistent recurrent kernel: one workgroup per direction (grid = 2).
// h-state in LDS (bf16), c-state in global f32, raw gate GEMM results staged
// in a per-direction global buffer between phase 1 (WMMA) and phase 2 (cell).
// ---------------------------------------------------------------------------
__global__ void __launch_bounds__(1024) k_recur(const __bf16* __restrict__ xproj,
                                                const __bf16* __restrict__ WhhF,
                                                const __bf16* __restrict__ WhhB,
                                                const float* __restrict__ bihF,
                                                const float* __restrict__ bhhF,
                                                const float* __restrict__ bihB,
                                                const float* __restrict__ bhhB,
                                                __bf16* __restrict__ xnext,
                                                float* __restrict__ gates,
                                                float* __restrict__ cstate) {
  __shared__ alignas(16) __bf16 hbuf[Bz * H];  // 48 KB
  const int dir = blockIdx.x;
  const __bf16* Whh = dir ? WhhB : WhhF;
  const float* bih = dir ? bihB : bihF;
  const float* bhh = dir ? bhhB : bhhF;
  const __bf16* xp0 = xproj + (size_t)dir * R * G4;
  float* gbuf = gates + (size_t)dir * Bz * G4;
  float* cst  = cstate + (size_t)dir * Bz * H;
  const int tid = threadIdx.x, lane = tid & 31, wave = tid >> 5;

  for (int i = tid; i < Bz * H; i += 1024) { hbuf[i] = (__bf16)0.0f; cst[i] = 0.0f; }
  __threadfence_block();
  __syncthreads();

  for (int step = 0; step < T; ++step) {
    const int t = dir ? (T - 1 - step) : step;
    const __bf16* xp = xp0 + (size_t)t * Bz * G4;

    // phase 1: gbuf[64,1536] = h @ Whh^T  (384 tiles, 12 per wave)
    for (int it = 0; it < 12; ++it) {
      const int tile = wave * 12 + it;
      const int m0 = (tile & 3) * 16;
      const int n0 = (tile >> 2) * 16;
      if (it + 1 < 12) {  // prefetch next tile's weight rows -> global_prefetch_b8
        const int nn = ((wave * 12 + it + 1) >> 2) * 16;
        __builtin_prefetch(Whh + (size_t)(nn + (lane & 15)) * H, 0, 3);
      }
      f32x8 c = {};
#pragma unroll
      for (int k0 = 0; k0 < H; k0 += 32) {
        bf16x16 a = load_frag(hbuf, H, m0, k0, lane);   // LDS reads
        bf16x16 b = load_frag(Whh,  H, n0, k0, lane);   // L2-resident weights
        c = wmma_bf16(a, b, c);
      }
      store_tile_f32(gbuf, G4, m0, n0, lane, c);
    }
    __threadfence_block();
    __syncthreads();

    // phase 2: LSTM cell, 24 elements per thread (torch gate order i,f,g,o)
    for (int e = 0; e < (Bz * H) / 1024; ++e) {
      const int lin = e * 1024 + tid;
      const int b_ = lin / H, j = lin % H;
      const int g0 = b_ * G4 + j;
      float gi = gbuf[g0]           + (float)xp[g0]           + bih[j]         + bhh[j];
      float gf = gbuf[g0 + H]       + (float)xp[g0 + H]       + bih[j + H]     + bhh[j + H];
      float gg = gbuf[g0 + 2 * H]   + (float)xp[g0 + 2 * H]   + bih[j + 2 * H] + bhh[j + 2 * H];
      float go = gbuf[g0 + 3 * H]   + (float)xp[g0 + 3 * H]   + bih[j + 3 * H] + bhh[j + 3 * H];
      float si = 1.0f / (1.0f + __expf(-gi));
      float sf = 1.0f / (1.0f + __expf(-gf));
      float so = 1.0f / (1.0f + __expf(-go));
      float cn = sf * cst[lin] + si * tanhf(gg);
      float hn = so * tanhf(cn);
      cst[lin] = cn;
      hbuf[lin] = (__bf16)hn;
      xnext[(size_t)(t * Bz + b_) * D + dir * H + j] = (__bf16)hn;  // concat(hf,hb)
    }
    __threadfence_block();
    __syncthreads();
  }
}

// ---------------------------------------------------------------------------
// FC (K=9) + LayerNorm. One wave per row; cross-lane reduce with shuffles.
// ---------------------------------------------------------------------------
__global__ void __launch_bounds__(256) k_fc_ln(const __bf16* __restrict__ feat,
                                               const float* __restrict__ fw,
                                               const float* __restrict__ fb,
                                               const float* __restrict__ lg,
                                               const float* __restrict__ lb,
                                               float* __restrict__ logits) {
  const int lane = threadIdx.x & 31, wave = threadIdx.x >> 5;
  const int row = blockIdx.x * 8 + wave;
  if (row >= R) return;
  const __bf16* f = feat + (size_t)row * D;
  float acc[K];
#pragma unroll
  for (int k = 0; k < K; ++k) acc[k] = 0.0f;
  for (int c = lane; c < D; c += 32) {
    float v = (float)f[c];
#pragma unroll
    for (int k = 0; k < K; ++k) acc[k] += v * fw[k * D + c];
  }
#pragma unroll
  for (int k = 0; k < K; ++k)
    for (int off = 16; off > 0; off >>= 1) acc[k] += __shfl_down(acc[k], off, 32);
  if (lane == 0) {
    float v[K], mu = 0.0f;
#pragma unroll
    for (int k = 0; k < K; ++k) { v[k] = acc[k] + fb[k]; mu += v[k]; }
    mu *= (1.0f / K);
    float var = 0.0f;
#pragma unroll
    for (int k = 0; k < K; ++k) { float d = v[k] - mu; var += d * d; }
    var *= (1.0f / K);
    float rs = rsqrtf(var + 1e-5f);
#pragma unroll
    for (int k = 0; k < K; ++k) logits[(size_t)row * K + k] = (v[k] - mu) * rs * lg[k] + lb[k];
  }
}

// ---------------------------------------------------------------------------
// CRF: forward log-partition + Viterbi + backtrack. 64 threads, lane = batch.
// ---------------------------------------------------------------------------
__global__ void __launch_bounds__(64) k_crf(const float* __restrict__ logits,
                                            const float* __restrict__ trans,
                                            float* __restrict__ score,
                                            int* __restrict__ path,
                                            int* __restrict__ psi) {
  __shared__ float tr[K * K];
  for (int i = threadIdx.x; i < K * K; i += 64) tr[i] = trans[i];
  __syncthreads();
  const int b = threadIdx.x;
  float la[K], ld[K];
#pragma unroll
  for (int k = 0; k < K; ++k) { la[k] = NEGV; ld[k] = NEGV; }
  la[START_TAG] = 0.0f;
  ld[START_TAG] = 0.0f;
  for (int t = 1; t < T; ++t) {
    const float* ft = logits + (size_t)(t * Bz + b) * K;
    float nla[K], nld[K];
#pragma unroll
    for (int j = 0; j < K; ++j) {
      float m = -3.4e38f;
#pragma unroll
      for (int i = 0; i < K; ++i) m = fmaxf(m, tr[j * K + i] + la[i]);
      float s = 0.0f;
#pragma unroll
      for (int i = 0; i < K; ++i) s += __expf(tr[j * K + i] + la[i] - m);
      nla[j] = m + __logf(s) + ft[j];
      float best = -3.4e38f; int bi = 0;
#pragma unroll
      for (int i = 0; i < K; ++i) {
        float v = tr[j * K + i] + ld[i];
        if (v > best) { best = v; bi = i; }
      }
      nld[j] = best + ft[j];
      psi[(size_t)((t - 1) * Bz + b) * K + j] = bi;
    }
#pragma unroll
    for (int k = 0; k < K; ++k) { la[k] = nla[k]; ld[k] = nld[k]; }
  }
  // score = logsumexp(la)
  float m = -3.4e38f;
#pragma unroll
  for (int k = 0; k < K; ++k) m = fmaxf(m, la[k]);
  float s = 0.0f;
#pragma unroll
  for (int k = 0; k < K; ++k) s += __expf(la[k] - m);
  score[b] = m + __logf(s);
  // backtrack
  float best = -3.4e38f; int p = 0;
#pragma unroll
  for (int k = 0; k < K; ++k) if (ld[k] > best) { best = ld[k]; p = k; }
  path[(size_t)b * T + (T - 1)] = p;
  for (int tt = T - 2; tt >= 0; --tt) {
    p = psi[(size_t)(tt * Bz + b) * K + p];
    path[(size_t)b * T + tt] = p;
  }
}

// ---------------------------------------------------------------------------
// Host orchestration
// ---------------------------------------------------------------------------
extern "C" void kernel_launch(void* const* d_in, const int* in_sizes, int n_in,
                              void* d_out, int out_size, void* d_ws, size_t ws_size,
                              hipStream_t stream) {
  (void)in_sizes; (void)n_in; (void)out_size; (void)ws_size;
  // Inputs (setup_inputs dict order, lstm_params flattened in insertion order):
  const float* embeds = (const float*)d_in[0];
  const float* Wih_in[4] = { (const float*)d_in[1], (const float*)d_in[5],
                             (const float*)d_in[9], (const float*)d_in[13] };  // l0f,l0b,l1f,l1b
  const float* Whh_in[4] = { (const float*)d_in[2], (const float*)d_in[6],
                             (const float*)d_in[10], (const float*)d_in[14] };
  const float* bih_in[4] = { (const float*)d_in[3], (const float*)d_in[7],
                             (const float*)d_in[11], (const float*)d_in[15] };
  const float* bhh_in[4] = { (const float*)d_in[4], (const float*)d_in[8],
                             (const float*)d_in[12], (const float*)d_in[16] };
  const float* fc_w = (const float*)d_in[17];
  const float* fc_b = (const float*)d_in[18];
  const float* ln_g = (const float*)d_in[19];
  const float* ln_b = (const float*)d_in[20];
  const float* trans = (const float*)d_in[21];

  // Workspace carve-out
  char* ws = (char*)d_ws;
  size_t off = 0;
  auto carve = [&](size_t bytes) -> char* {
    char* p = ws + off;
    off = (off + bytes + 255) & ~(size_t)255;
    return p;
  };
  __bf16* xA     = (__bf16*)carve((size_t)R * D * 2);           // layer0 in / layer1 out (feats)
  __bf16* xB     = (__bf16*)carve((size_t)R * D * 2);           // layer0 out / layer1 in
  __bf16* xproj  = (__bf16*)carve((size_t)2 * R * G4 * 2);      // per-dir input projections
  __bf16* wihbf[4]; for (int i = 0; i < 4; ++i) wihbf[i] = (__bf16*)carve((size_t)G4 * D * 2);
  __bf16* whhbf[4]; for (int i = 0; i < 4; ++i) whhbf[i] = (__bf16*)carve((size_t)G4 * H * 2);
  float*  gates  = (float*)carve((size_t)2 * Bz * G4 * 4);
  float*  cstate = (float*)carve((size_t)2 * Bz * H * 4);
  float*  logits = (float*)carve((size_t)R * K * 4);
  int*    psi    = (int*)carve((size_t)(T - 1) * Bz * K * 4);

  float* score = (float*)d_out;
  int*   path  = (int*)d_out + Bz;

  // Conversions
  {
    int n = Bz * T * D;
    k_embed_tr<<<(n + 255) / 256, 256, 0, stream>>>(embeds, xA);
    for (int i = 0; i < 4; ++i) {
      int ni = G4 * D;
      k_cvt_bf16<<<(ni + 255) / 256, 256, 0, stream>>>(Wih_in[i], wihbf[i], ni);
      int nh = G4 * H;
      k_cvt_bf16<<<(nh + 255) / 256, 256, 0, stream>>>(Whh_in[i], whhbf[i], nh);
    }
  }

  dim3 gx(12, 512, 2);
  // Layer 0
  k_xproj<<<gx, 256, 0, stream>>>(xA, wihbf[0], wihbf[1], xproj);
  k_recur<<<2, 1024, 0, stream>>>(xproj, whhbf[0], whhbf[1],
                                  bih_in[0], bhh_in[0], bih_in[1], bhh_in[1],
                                  xB, gates, cstate);
  // Layer 1
  k_xproj<<<gx, 256, 0, stream>>>(xB, wihbf[2], wihbf[3], xproj);
  k_recur<<<2, 1024, 0, stream>>>(xproj, whhbf[2], whhbf[3],
                                  bih_in[2], bhh_in[2], bih_in[3], bhh_in[3],
                                  xA, gates, cstate);
  // FC + LayerNorm
  k_fc_ln<<<R / 8, 256, 0, stream>>>(xA, fc_w, fc_b, ln_g, ln_b, logits);
  // CRF forward + Viterbi
  k_crf<<<1, 64, 0, stream>>>(logits, trans, score, path, psi);
}
